// SAM2MultiScaleBlock_55405078119085
// MI455X (gfx1250) — compile-verified, matrix operand order checked by reading, not run on hardware
//
#include <hip/hip_runtime.h>
#include <hip/hip_bf16.h>
#include <math.h>
#include <stdint.h>

// ---------------------------------------------------------------------------
// SAM2 MultiScaleBlock for MI455X (gfx1250, wave32, WMMA f16 -> f32 accum)
// B=8 H=W=64 DIM=448 DIM_OUT=896 HEADS=16 HD=56 WIN=14 STRIDE=2 MLP=3584
// GEMMs: double-buffered LDS fed by global_load_async_to_lds_b128 (ASYNCcnt),
// weights pre-transposed to [N,K] f16 so A and B tiles load identically.
// ---------------------------------------------------------------------------

typedef _Float16 half_t;
typedef __attribute__((ext_vector_type(16))) _Float16 v16h;
typedef __attribute__((ext_vector_type(8)))  _Float16 v8h;
typedef __attribute__((ext_vector_type(8)))  float    v8f;

static constexpr int Bc = 8, Hc = 64, Wc = 64;
static constexpr int DIM = 448, DIM_OUT = 896, HEADS = 16, HD = 56;
static constexpr int WIN = 14, NH = 5, NW = 5;           // padded 70x70 -> 5x5 windows
static constexpr int BW = Bc * NH * NW;                  // 200 windows
static constexpr int TOK = WIN * WIN;                    // 196 tokens / window
static constexpr int WQ = 7, NQ = WQ * WQ;               // 49 pooled queries
static constexpr int MLP_DIM = 3584;
static constexpr int H2 = 32, W2 = 32;                   // pooled spatial
static constexpr int NPIX = Bc * Hc * Wc;                // 32768
static constexpr int NPIX2 = Bc * H2 * W2;               // 8192
static constexpr float ATTN_SCALE = 0.13363062095621219f; // 1/sqrt(56)

// ------------------------------- utility kernels ---------------------------

// src [Kd, Nd] fp32 -> dst [Nd, Kd] f16 (weights, done once; tiny).
__global__ void f32_to_f16_transpose_kernel(const float* __restrict__ src,
                                            half_t* __restrict__ dst,
                                            int Kd, int Nd) {
  int i = blockIdx.x * blockDim.x + threadIdx.x;
  if (i >= Kd * Nd) return;
  const int k = i / Nd, n = i % Nd;
  dst[(size_t)n * Kd + k] = (half_t)src[i];
}

// LayerNorm over last dim C, one block per row, output f16.
__global__ __launch_bounds__(256) void ln_f16_kernel(
    const float* __restrict__ x, const float* __restrict__ g,
    const float* __restrict__ b, half_t* __restrict__ out, int C) {
  const int row = blockIdx.x;
  const float* xr = x + (size_t)row * C;
  __shared__ float red0[8], red1[8];
  float s = 0.f, s2 = 0.f;
  for (int c = threadIdx.x; c < C; c += blockDim.x) {
    float v = xr[c]; s += v; s2 += v * v;
  }
  for (int m = 16; m >= 1; m >>= 1) {
    s  += __shfl_xor(s,  m, 32);
    s2 += __shfl_xor(s2, m, 32);
  }
  const int wave = threadIdx.x >> 5;
  if ((threadIdx.x & 31) == 0) { red0[wave] = s; red1[wave] = s2; }
  __syncthreads();
  float ts = 0.f, ts2 = 0.f;
  for (int w = 0; w < 8; ++w) { ts += red0[w]; ts2 += red1[w]; }
  const float mean = ts / (float)C;
  const float var  = ts2 / (float)C - mean * mean;
  const float rs   = rsqrtf(var + 1e-6f);
  half_t* orow = out + (size_t)row * C;
  for (int c = threadIdx.x; c < C; c += blockDim.x)
    orow[c] = (half_t)((xr[c] - mean) * rs * g[c] + b[c]);
}

// Gather hn into padded windows: win[wb, t, :DIM] (zero pad beyond 64x64).
__global__ void win_gather_kernel(const half_t* __restrict__ hn,
                                  half_t* __restrict__ win) {
  const int row = blockIdx.x;               // wb*196 + t
  const int wb = row / TOK, t = row % TOK;
  const int b = wb / (NH * NW);
  const int wrem = wb % (NH * NW);
  const int wh = wrem / NW, ww = wrem % NW;
  const int py = wh * WIN + t / WIN;
  const int px = ww * WIN + t % WIN;
  const int c = threadIdx.x;                // 0..63, use <56 (56*8 = 448)
  if (c >= DIM / 8) return;
  int4 val = make_int4(0, 0, 0, 0);
  if (py < Hc && px < Wc)
    val = *(const int4*)&hn[(((size_t)b * Hc + py) * Wc + px) * DIM + c * 8];
  *(int4*)&win[(size_t)row * DIM + c * 8] = val;
}

// 2x2 maxpool, fp32, [B,64,64,C] -> [B,32,32,C]
__global__ void maxpool2x2_f32_kernel(const float* __restrict__ in,
                                      float* __restrict__ out, int C) {
  const int row = blockIdx.x;               // b*1024 + y*32 + x
  const int b = row >> 10, rr = row & 1023;
  const int y = rr >> 5, x = rr & 31;
  const float* p00 = in + (((size_t)(b * Hc + 2 * y)) * Wc + 2 * x) * C;
  const float* p01 = p00 + C;
  const float* p10 = p00 + (size_t)Wc * C;
  const float* p11 = p10 + C;
  float* o = out + (size_t)row * C;
  for (int c = threadIdx.x; c < C; c += blockDim.x)
    o[c] = fmaxf(fmaxf(p00[c], p01[c]), fmaxf(p10[c], p11[c]));
}

// q 2x2 maxpool from fused qkv (q = cols [0,896)), f16 in/out.
__global__ void qpool_kernel(const half_t* __restrict__ qkv,
                             half_t* __restrict__ qp) {
  const int row = blockIdx.x;               // wb*49 + qy*7+qx
  const int wb = row / NQ, q = row % NQ;
  const int qy = q / WQ, qx = q % WQ;
  const half_t* base = qkv + (size_t)wb * TOK * (3 * DIM_OUT);
  const int t00 = (2 * qy) * WIN + 2 * qx;
  const size_t ld = 3 * DIM_OUT;
  for (int c = threadIdx.x; c < DIM_OUT; c += blockDim.x) {
    float v0 = (float)base[(size_t)t00 * ld + c];
    float v1 = (float)base[(size_t)(t00 + 1) * ld + c];
    float v2 = (float)base[(size_t)(t00 + WIN) * ld + c];
    float v3 = (float)base[(size_t)(t00 + WIN + 1) * ld + c];
    qp[(size_t)row * DIM_OUT + c] = (half_t)fmaxf(fmaxf(v0, v1), fmaxf(v2, v3));
  }
}

// Attention: one block per (window, head). K/V tile in LDS, fp32 softmax.
__global__ __launch_bounds__(256) void attn_kernel(
    const half_t* __restrict__ qkv, const half_t* __restrict__ qp,
    half_t* __restrict__ out) {
  const int wb = blockIdx.x;
  const int h  = blockIdx.y;
  __shared__ half_t sk[TOK * HD];
  __shared__ half_t sv[TOK * HD];
  __shared__ float  sq[HD];
  __shared__ float  sc[TOK];
  __shared__ float  red[8];
  __shared__ float  pd[4 * HD];
  const int tid = threadIdx.x;
  const int wave = tid >> 5, lane = tid & 31;
  const size_t ld = 3 * DIM_OUT;
  const half_t* kbase = qkv + (size_t)wb * TOK * ld + DIM_OUT + h * HD;
  const half_t* vbase = kbase + DIM_OUT;
  for (int i = tid; i < TOK * HD; i += 256) {
    const int t = i / HD, d = i % HD;
    sk[i] = kbase[(size_t)t * ld + d];
    sv[i] = vbase[(size_t)t * ld + d];
  }
  __syncthreads();
  for (int qi = 0; qi < NQ; ++qi) {
    if (tid < HD)
      sq[tid] = (float)qp[((size_t)wb * NQ + qi) * DIM_OUT + h * HD + tid] * ATTN_SCALE;
    __syncthreads();
    if (tid < TOK) {
      float dot = 0.f;
      for (int d = 0; d < HD; ++d) dot += sq[d] * (float)sk[tid * HD + d];
      sc[tid] = dot;
    }
    __syncthreads();
    float myv = (tid < TOK) ? sc[tid] : -1e30f;
    float m = myv;
    for (int s = 16; s >= 1; s >>= 1) m = fmaxf(m, __shfl_xor(m, s, 32));
    if (lane == 0) red[wave] = m;
    __syncthreads();
    float mx = red[0];
    for (int w = 1; w < 8; ++w) mx = fmaxf(mx, red[w]);
    const float e = (tid < TOK) ? __expf(myv - mx) : 0.f;
    float s = e;
    __syncthreads();           // red reuse
    for (int sh = 16; sh >= 1; sh >>= 1) s += __shfl_xor(s, sh, 32);
    if (lane == 0) red[wave] = s;
    __syncthreads();
    float tot = 0.f;
    for (int w = 0; w < 8; ++w) tot += red[w];
    if (tid < TOK) sc[tid] = e / tot;
    __syncthreads();
    if (tid < 4 * HD) {
      const int d = tid % HD, ch = tid / HD;   // 196 = 4*49
      float a = 0.f;
      for (int t = ch * 49; t < ch * 49 + 49; ++t)
        a += sc[t] * (float)sv[t * HD + d];
      pd[tid] = a;
    }
    __syncthreads();
    if (tid < HD) {
      const float a = pd[tid] + pd[HD + tid] + pd[2 * HD + tid] + pd[3 * HD + tid];
      out[((size_t)wb * NQ + qi) * DIM_OUT + h * HD + tid] = (half_t)a;
    }
    __syncthreads();
  }
}

// Un-window (crop 35->32), add shortcut -> y fp32 [8192, 896]
__global__ void assemble_y_kernel(const float* __restrict__ shortcut,
                                  const float* __restrict__ projout,
                                  float* __restrict__ y) {
  const int row = blockIdx.x;               // b*1024 + r*32 + c
  const int b = row >> 10, rr = row & 1023;
  const int r = rr >> 5, c = rr & 31;
  const int wh = r / WQ, qy = r % WQ, ww = c / WQ, qx = c % WQ;
  const size_t src = ((size_t)((b * NH + wh) * NW + ww) * NQ + qy * WQ + qx) * DIM_OUT;
  const size_t dst = (size_t)row * DIM_OUT;
  for (int d = threadIdx.x; d < DIM_OUT; d += blockDim.x)
    y[dst + d] = shortcut[dst + d] + projout[src + d];
}

// ------------------------------- WMMA GEMM ---------------------------------
// C[M,N] = A[M,K](f16) @ Bt[N,K](f16, pre-transposed weights) + bias.
// 128x128x64 tiles, 8 waves (4x2), double-buffered LDS filled with
// global_load_async_to_lds_b128; 16 v_wmma_f32_16x16x32_f16 per K-tile/wave.
// EPI: 0 = fp32 out, 1 = f16 out, 2 = gelu -> f16 out, 3 = fp32 out + addsrc.

#define BM 128
#define BN 128
#define BK 64
#define LDSS 72   // BK + 8 halves -> 144B rows, keeps 16B chunk alignment

__device__ __forceinline__ void async_load_b128(uint32_t lds_off,
                                                const half_t* gaddr) {
  asm volatile("global_load_async_to_lds_b128 %0, %1, off"
               :: "v"(lds_off), "v"((uint64_t)(uintptr_t)gaddr)
               : "memory");
}
__device__ __forceinline__ void wait_asynccnt0() {
  asm volatile("s_wait_asynccnt 0x0" ::: "memory");
}

template <int EPI>
__global__ __launch_bounds__(256, 2) void gemm_f16_kernel(
    const half_t* __restrict__ A, const half_t* __restrict__ Bt,
    const float* __restrict__ bias, float* __restrict__ outF,
    half_t* __restrict__ outH, const float* __restrict__ addsrc,
    int M, int N, int K) {
  __shared__ alignas(16) half_t sA[2][BM * LDSS];
  __shared__ alignas(16) half_t sB[2][BN * LDSS];

  const int tid  = threadIdx.x;
  const int wave = tid >> 5;
  const int lane = tid & 31;
  const int l16  = lane & 15;
  const int hs   = lane >> 4;        // half-select (lanes 0-15 / 16-31)
  const int wm   = wave >> 1;        // 0..3 -> rows wm*32
  const int wn   = wave & 1;         // 0..1 -> cols wn*64
  const int m0   = blockIdx.y * BM;
  const int n0   = blockIdx.x * BN;

  v8f acc[2][4];
  #pragma unroll
  for (int mi = 0; mi < 2; ++mi)
    #pragma unroll
    for (int ni = 0; ni < 4; ++ni)
      #pragma unroll
      for (int r = 0; r < 8; ++r) acc[mi][ni][r] = 0.f;

  const int T = K / BK;

  // Issue one K-tile worth of async loads (8 x b128 per thread) into buf.
  auto issue = [&](int kt, int buf) {
    const int k0 = kt * BK;
    #pragma unroll
    for (int i = 0; i < 4; ++i) {          // A: 1024 chunks of 8 halves
      const int c = tid + i * 256;
      const int row = c >> 3, q = c & 7;
      int gm = m0 + row; if (gm >= M) gm = M - 1;   // clamp: rows >= M unused
      async_load_b128((uint32_t)(uintptr_t)&sA[buf][row * LDSS + q * 8],
                      &A[(size_t)gm * K + k0 + q * 8]);
    }
    #pragma unroll
    for (int i = 0; i < 4; ++i) {          // Bt: same pattern along K
      const int c = tid + i * 256;
      const int row = c >> 3, q = c & 7;
      async_load_b128((uint32_t)(uintptr_t)&sB[buf][row * LDSS + q * 8],
                      &Bt[(size_t)(n0 + row) * K + k0 + q * 8]);
    }
  };

  issue(0, 0);
  wait_asynccnt0();
  __syncthreads();

  for (int kt = 0; kt < T; ++kt) {
    const int cur = kt & 1;
    if (kt + 1 < T) issue(kt + 1, cur ^ 1);

    #pragma unroll
    for (int s = 0; s < 2; ++s) {          // two k=32 sub-steps per tile
      v16h afrag[2], bfrag[4];
      #pragma unroll
      for (int mi = 0; mi < 2; ++mi) {
        const int mrow = wm * 32 + mi * 16 + l16;
        const v8h lo = *(const v8h*)&sA[cur][mrow * LDSS + s * 32 + hs * 8];
        const v8h hi = *(const v8h*)&sA[cur][mrow * LDSS + s * 32 + hs * 8 + 16];
        #pragma unroll
        for (int e = 0; e < 8; ++e) { afrag[mi][e] = lo[e]; afrag[mi][e + 8] = hi[e]; }
      }
      #pragma unroll
      for (int ni = 0; ni < 4; ++ni) {
        const int ncol = wn * 64 + ni * 16 + l16;
        const v8h lo = *(const v8h*)&sB[cur][ncol * LDSS + s * 32 + hs * 8];
        const v8h hi = *(const v8h*)&sB[cur][ncol * LDSS + s * 32 + hs * 8 + 16];
        #pragma unroll
        for (int e = 0; e < 8; ++e) { bfrag[ni][e] = lo[e]; bfrag[ni][e + 8] = hi[e]; }
      }
      #pragma unroll
      for (int mi = 0; mi < 2; ++mi)
        #pragma unroll
        for (int ni = 0; ni < 4; ++ni)
          acc[mi][ni] = __builtin_amdgcn_wmma_f32_16x16x32_f16(
              false, afrag[mi], false, bfrag[ni], (short)0, acc[mi][ni],
              false, false);
    }

    if (kt + 1 < T) wait_asynccnt0();   // next tile resident before barrier
    __syncthreads();                    // nobody still reading buf we overwrite
  }

  // Epilogue: C layout — col = l16, row = r + 8*hs within each 16x16 tile.
  #pragma unroll
  for (int mi = 0; mi < 2; ++mi)
    #pragma unroll
    for (int ni = 0; ni < 4; ++ni) {
      const int gn = n0 + wn * 64 + ni * 16 + l16;
      const float bv = bias ? bias[gn] : 0.f;
      #pragma unroll
      for (int r = 0; r < 8; ++r) {
        const int gm = m0 + wm * 32 + mi * 16 + r + hs * 8;
        if (gm >= M) continue;
        float v = acc[mi][ni][r] + bv;
        const size_t idx = (size_t)gm * N + gn;
        if constexpr (EPI == 0) {
          outF[idx] = v;
        } else if constexpr (EPI == 1) {
          outH[idx] = (half_t)v;
        } else if constexpr (EPI == 2) {
          v = 0.5f * v * (1.f + erff(v * 0.70710678118654752f));
          outH[idx] = (half_t)v;
        } else {
          outF[idx] = v + addsrc[idx];
        }
      }
    }
}

// ------------------------------- launcher ----------------------------------

extern "C" void kernel_launch(void* const* d_in, const int* in_sizes, int n_in,
                              void* d_out, int out_size, void* d_ws, size_t ws_size,
                              hipStream_t stream) {
  const float* x        = (const float*)d_in[0];
  const float* ln1_g    = (const float*)d_in[1];
  const float* ln1_b    = (const float*)d_in[2];
  const float* qkv_w    = (const float*)d_in[3];
  const float* qkv_b    = (const float*)d_in[4];
  const float* proj_w   = (const float*)d_in[5];
  const float* proj_b   = (const float*)d_in[6];
  const float* skip_w   = (const float*)d_in[7];
  const float* skip_b   = (const float*)d_in[8];
  const float* ln2_g    = (const float*)d_in[9];
  const float* ln2_b    = (const float*)d_in[10];
  const float* mlp1_w   = (const float*)d_in[11];
  const float* mlp1_b   = (const float*)d_in[12];
  const float* mlp2_w   = (const float*)d_in[13];
  const float* mlp2_b   = (const float*)d_in[14];
  float* out = (float*)d_out;

  // Bump allocator over d_ws (256B aligned).
  char* ws = (char*)d_ws;
  size_t off = 0;
  auto alloc = [&](size_t bytes) -> void* {
    void* p = ws + off;
    off = (off + bytes + 255) & ~(size_t)255;
    return p;
  };
  half_t* hn       = (half_t*)alloc((size_t)NPIX * DIM * 2);
  half_t* w_qkv    = (half_t*)alloc((size_t)DIM * 3 * DIM_OUT * 2);   // [2688,448]
  half_t* w_skip   = (half_t*)alloc((size_t)DIM * DIM_OUT * 2);       // [896,448]
  half_t* w_proj   = (half_t*)alloc((size_t)DIM_OUT * DIM_OUT * 2);   // [896,896]
  half_t* w_mlp1   = (half_t*)alloc((size_t)DIM_OUT * MLP_DIM * 2);   // [3584,896]
  half_t* w_mlp2   = (half_t*)alloc((size_t)MLP_DIM * DIM_OUT * 2);   // [896,3584]
  half_t* win      = (half_t*)alloc((size_t)BW * TOK * DIM * 2);
  float*  skipfull = (float*) alloc((size_t)NPIX * DIM_OUT * 4);
  float*  shortcut = (float*) alloc((size_t)NPIX2 * DIM_OUT * 4);
  half_t* qkv      = (half_t*)alloc((size_t)BW * TOK * 3 * DIM_OUT * 2);
  half_t* qp       = (half_t*)alloc((size_t)BW * NQ * DIM_OUT * 2);
  half_t* attnout  = (half_t*)alloc((size_t)BW * NQ * DIM_OUT * 2);
  float*  projout  = (float*) alloc((size_t)BW * NQ * DIM_OUT * 4);
  float*  y        = (float*) alloc((size_t)NPIX2 * DIM_OUT * 4);
  half_t* hn2      = (half_t*)alloc((size_t)NPIX2 * DIM_OUT * 2);
  half_t* mlp1o    = (half_t*)alloc((size_t)NPIX2 * MLP_DIM * 2);
  (void)ws_size; (void)n_in; (void)in_sizes; (void)out_size;

  // Weight conversions: fp32 [K,N] -> f16 [N,K].
  auto cvtT = [&](const float* s, half_t* d, int Kd, int Nd) {
    const int n = Kd * Nd;
    f32_to_f16_transpose_kernel<<<(n + 255) / 256, 256, 0, stream>>>(s, d, Kd, Nd);
  };
  cvtT(qkv_w,  w_qkv,  DIM, 3 * DIM_OUT);
  cvtT(skip_w, w_skip, DIM, DIM_OUT);
  cvtT(proj_w, w_proj, DIM_OUT, DIM_OUT);
  cvtT(mlp1_w, w_mlp1, DIM_OUT, MLP_DIM);
  cvtT(mlp2_w, w_mlp2, MLP_DIM, DIM_OUT);

  // LN1 -> hn (f16)
  ln_f16_kernel<<<NPIX, 256, 0, stream>>>(x, ln1_g, ln1_b, hn, DIM);

  // skip = maxpool(hn @ skip_w + b)
  gemm_f16_kernel<0><<<dim3(DIM_OUT / BN, NPIX / BM), 256, 0, stream>>>(
      hn, w_skip, skip_b, skipfull, nullptr, nullptr, NPIX, DIM_OUT, DIM);
  maxpool2x2_f32_kernel<<<NPIX2, 256, 0, stream>>>(skipfull, shortcut, DIM_OUT);

  // windows + fused qkv GEMM
  win_gather_kernel<<<BW * TOK, 64, 0, stream>>>(hn, win);
  gemm_f16_kernel<1><<<dim3(3 * DIM_OUT / BN, (BW * TOK + BM - 1) / BM), 256, 0,
                       stream>>>(win, w_qkv, qkv_b, nullptr, qkv, nullptr,
                                 BW * TOK, 3 * DIM_OUT, DIM);

  // pooled q, attention
  qpool_kernel<<<BW * NQ, 256, 0, stream>>>(qkv, qp);
  attn_kernel<<<dim3(BW, HEADS), 256, 0, stream>>>(qkv, qp, attnout);

  // output projection, un-window + shortcut
  gemm_f16_kernel<0><<<dim3(DIM_OUT / BN, (BW * NQ + BM - 1) / BM), 256, 0,
                       stream>>>(attnout, w_proj, proj_b, projout, nullptr,
                                 nullptr, BW * NQ, DIM_OUT, DIM_OUT);
  assemble_y_kernel<<<NPIX2, 256, 0, stream>>>(shortcut, projout, y);

  // MLP with residual
  ln_f16_kernel<<<NPIX2, 256, 0, stream>>>(y, ln2_g, ln2_b, hn2, DIM_OUT);
  gemm_f16_kernel<2><<<dim3(MLP_DIM / BN, NPIX2 / BM), 256, 0, stream>>>(
      hn2, w_mlp1, mlp1_b, nullptr, mlp1o, nullptr, NPIX2, MLP_DIM, DIM_OUT);
  gemm_f16_kernel<3><<<dim3(DIM_OUT / BN, NPIX2 / BM), 256, 0, stream>>>(
      mlp1o, w_mlp2, mlp2_b, out, nullptr, y, NPIX2, DIM_OUT, MLP_DIM);
}
